// ContextAwareAttention_8057358648291
// MI455X (gfx1250) — matrix operand order repaired and checked by reading, not compiled, not run on hardware
//
#include <hip/hip_runtime.h>
#include <hip/hip_bf16.h>

// ---------------------------------------------------------------------------
// ContextAwareAttention, fused for MI455X (gfx1250, wave32).
//   B=16, S=4096, D=1024.
// Pipeline:
//   k0: W_in fp32 -> bf16 (2MB, stays L2-resident for whole dispatch)
//   k1: pc[b,e] = context[b]·W_ctx[e] + b_ctx[e] + b_in[e]     (tiny)
//   k2: scores[b,s] = w_att · tanh(x[b,s]·W_inT + pc[b]) + b_att
//       -> bf16 V_WMMA_F32_16X16X32_BF16 core, f32 accumulate,
//          tanh + w_att reduction fused in epilogue (pi never stored)
//   k3: softmax over S per batch
//   k4: partial[b,d] += weights[b,s]*x[b,s,d]  (S split 16 ways, no atomics)
//   k5: reduce partials -> out[b,d]
// ---------------------------------------------------------------------------

#define DD 1024
#define SS 4096
#define BB 16

typedef __attribute__((ext_vector_type(16))) __bf16 v16bf;
typedef __attribute__((ext_vector_type(8)))  float  v8f;

union ABu {            // one WMMA A or B bf16 fragment: 16 bf16 = 8 VGPRs
    v16bf v;
    unsigned int u[8];
    uint4 q[2];
};
static_assert(sizeof(ABu) == 32, "fragment size");

// ---- k0: convert W_in to bf16 ---------------------------------------------
__global__ void convW_kernel(const float* __restrict__ W, __bf16* __restrict__ Wbf, int n) {
    int i = blockIdx.x * blockDim.x + threadIdx.x;
    if (i < n) Wbf[i] = (__bf16)W[i];
}

// ---- k1: pc[b,e] = context[b]·W_ctx[e] + b_ctx[e] + b_in[e] ----------------
__global__ void pc_kernel(const float* __restrict__ ctx, const float* __restrict__ Wctx,
                          const float* __restrict__ b_ctx, const float* __restrict__ b_in,
                          float* __restrict__ pc) {
    int gid = blockIdx.x * blockDim.x + threadIdx.x;   // BB*DD threads
    int b = gid >> 10, e = gid & (DD - 1);
    const float* c = ctx + b * DD;
    const float* w = Wctx + (size_t)e * DD;
    float s = 0.f;
    for (int d = 0; d < DD; d += 4) {
        float4 cv = *(const float4*)&c[d];
        float4 wv = *(const float4*)&w[d];
        s += cv.x * wv.x + cv.y * wv.y + cv.z * wv.z + cv.w * wv.w;
    }
    pc[gid] = s + b_ctx[e] + b_in[e];
}

// ---- k2: fused score GEMM --------------------------------------------------
// block = 256 threads = 8 waves; block covers 32 s-rows (2 WMMA M-tiles).
// wave w covers e-range [w*128, w*128+128) = 8 WMMA N-tiles (all waves share A).
__global__ __launch_bounds__(256)
void scores_kernel(const float* __restrict__ x, const __bf16* __restrict__ Wbf,
                   const float* __restrict__ pc, const float* __restrict__ w_att,
                   const float* __restrict__ b_att, float* __restrict__ scores) {
    const int b    = blockIdx.y;
    const int s0   = blockIdx.x * 32;
    const int tid  = threadIdx.x;
    const int wave = tid >> 5;
    const int lane = tid & 31;
    const int laneN = lane & 15;
    const int half  = lane >> 4;

    __shared__ __bf16 Atile[32][36];   // 32 rows x 32 K, +4 pad (bank spread)
    __shared__ float  Spart[8][32];    // per-wave partial row sums

    v8f acc[2][8] = {};                // 2 M-tiles x 8 N-tiles, f32 accum

    for (int k0 = 0; k0 < DD; k0 += 32) {
        // stage A: 32 rows x 32 K of x, fp32 -> bf16 (coalesced float4 loads)
        {
            int m  = tid >> 3;                // 0..31
            int kk = (tid & 7) * 4;           // 0,4,...,28
            const float4 xa = *(const float4*)&x[((size_t)(b * SS + s0 + m)) * DD + k0 + kk];
            Atile[m][kk + 0] = (__bf16)xa.x;
            Atile[m][kk + 1] = (__bf16)xa.y;
            Atile[m][kk + 2] = (__bf16)xa.z;
            Atile[m][kk + 3] = (__bf16)xa.w;
        }
        __syncthreads();

        // A fragments (contiguous K order, element i -> K = 16*half + i, same as B below).
        // lane: M = lane%16; element pair j holds K = {base, base+1}.
        ABu a[2];
        #pragma unroll
        for (int m = 0; m < 2; m++) {
            int row = m * 16 + laneN;
            #pragma unroll
            for (int j = 0; j < 8; j++) {
                int kb = half * 16 + 2 * j;
                a[m].u[j] = *(const unsigned int*)&Atile[row][kb];
            }
        }

        // B fragments straight from global bf16 W_in (L2-resident):
        // lane: N = lane%16, elements i -> K = 16*half + i (contiguous bytes).
        #pragma unroll
        for (int t = 0; t < 8; t++) {
            const int e0 = wave * 128 + t * 16;
            ABu bf;
            const uint4* bp = (const uint4*)(Wbf + (size_t)(e0 + laneN) * DD + k0 + 16 * half);
            bf.q[0] = bp[0];
            bf.q[1] = bp[1];
            acc[0][t] = __builtin_amdgcn_wmma_f32_16x16x32_bf16(
                false, a[0].v, false, bf.v, (short)0, acc[0][t], false, false);
            acc[1][t] = __builtin_amdgcn_wmma_f32_16x16x32_bf16(
                false, a[1].v, false, bf.v, (short)0, acc[1][t], false, false);
        }
        __syncthreads();
    }

    // Epilogue: per element pi -> tanh(pi + pc)*w_att, then reduce over e.
    // C/D layout: lane col N=lane%16, VGPR r -> row r + 8*half (+16*m).
    float rs[2][8];
    #pragma unroll
    for (int m = 0; m < 2; m++)
        #pragma unroll
        for (int r = 0; r < 8; r++) rs[m][r] = 0.f;

    #pragma unroll
    for (int t = 0; t < 8; t++) {
        const int e   = wave * 128 + t * 16 + laneN;
        const float pcv = pc[b * DD + e];
        const float wv  = w_att[e];
        #pragma unroll
        for (int m = 0; m < 2; m++)
            #pragma unroll
            for (int r = 0; r < 8; r++)
                rs[m][r] += tanhf(acc[m][t][r] + pcv) * wv;
    }
    // reduce over the 16 lanes of each half (N dimension)
    #pragma unroll
    for (int m = 0; m < 2; m++)
        #pragma unroll
        for (int r = 0; r < 8; r++) {
            float v = rs[m][r];
            v += __shfl_xor(v, 1, 32);
            v += __shfl_xor(v, 2, 32);
            v += __shfl_xor(v, 4, 32);
            v += __shfl_xor(v, 8, 32);
            rs[m][r] = v;
        }
    if (laneN == 0) {
        #pragma unroll
        for (int m = 0; m < 2; m++)
            #pragma unroll
            for (int r = 0; r < 8; r++)
                Spart[wave][m * 16 + r + 8 * half] = rs[m][r];
    }
    __syncthreads();
    if (tid < 32) {
        float sum = 0.f;
        #pragma unroll
        for (int w = 0; w < 8; w++) sum += Spart[w][tid];
        scores[b * SS + s0 + tid] = sum + b_att[0];
    }
}

// ---- k3: softmax over S per batch -----------------------------------------
__global__ __launch_bounds__(256)
void softmax_kernel(const float* __restrict__ scores, float* __restrict__ weights) {
    const int b = blockIdx.x, tid = threadIdx.x;
    __shared__ float red[256];
    float lmax = -INFINITY;
    for (int i = tid; i < SS; i += 256) lmax = fmaxf(lmax, scores[b * SS + i]);
    red[tid] = lmax; __syncthreads();
    for (int s = 128; s > 0; s >>= 1) { if (tid < s) red[tid] = fmaxf(red[tid], red[tid + s]); __syncthreads(); }
    const float m = red[0]; __syncthreads();
    float lsum = 0.f;
    for (int i = tid; i < SS; i += 256) {
        float e = __expf(scores[b * SS + i] - m);
        weights[b * SS + i] = e;
        lsum += e;
    }
    red[tid] = lsum; __syncthreads();
    for (int s = 128; s > 0; s >>= 1) { if (tid < s) red[tid] += red[tid + s]; __syncthreads(); }
    const float inv = 1.0f / red[0];
    for (int i = tid; i < SS; i += 256) weights[b * SS + i] *= inv;
}

// ---- k4: partial weighted sum over an S-chunk (deterministic, no atomics) --
__global__ __launch_bounds__(256)
void wsum_kernel(const float* __restrict__ x, const float* __restrict__ weights,
                 float* __restrict__ partial) {
    const int dc = blockIdx.x, sc = blockIdx.y, b = blockIdx.z;
    const int tid = threadIdx.x;
    __shared__ float wsh[256];
    wsh[tid] = weights[b * SS + sc * 256 + tid];
    __syncthreads();
    const int d = dc * 256 + tid;
    const float* xp = x + ((size_t)(b * SS + sc * 256)) * DD + d;
    float acc = 0.f;
    for (int sl = 0; sl < 256; ++sl)
        acc += wsh[sl] * xp[(size_t)sl * DD];
    partial[(size_t)(sc * BB + b) * DD + d] = acc;
}

// ---- k5: reduce S-chunk partials ------------------------------------------
__global__ void reduce_kernel(const float* __restrict__ partial, float* __restrict__ out) {
    const int i = blockIdx.x * 256 + threadIdx.x;   // BB*DD
    float s = 0.f;
    for (int sc = 0; sc < 16; sc++) s += partial[(size_t)sc * BB * DD + i];
    out[i] = s;
}

extern "C" void kernel_launch(void* const* d_in, const int* in_sizes, int n_in,
                              void* d_out, int out_size, void* d_ws, size_t ws_size,
                              hipStream_t stream) {
    const float* x       = (const float*)d_in[0];
    const float* context = (const float*)d_in[1];
    const float* W_in    = (const float*)d_in[2];
    const float* b_in    = (const float*)d_in[3];
    const float* W_ctx   = (const float*)d_in[4];
    const float* b_ctx   = (const float*)d_in[5];
    const float* w_att   = (const float*)d_in[6];
    const float* b_att   = (const float*)d_in[7];
    float* out = (float*)d_out;

    // workspace layout (≈3.6 MB total)
    char* ws = (char*)d_ws;
    __bf16* Wbf    = (__bf16*)ws;                                   // 2 MB
    float*  pc     = (float*)(ws + (size_t)(2 << 20));              // 64 KB
    float*  scores = pc + BB * DD;                                  // 256 KB
    float*  weights = scores + BB * SS;                             // 256 KB
    float*  partial = weights + BB * SS;                            // 1 MB

    convW_kernel<<<(DD * DD) / 256, 256, 0, stream>>>(W_in, Wbf, DD * DD);
    pc_kernel<<<(BB * DD) / 256, 256, 0, stream>>>(context, W_ctx, b_ctx, b_in, pc);
    scores_kernel<<<dim3(SS / 32, BB), 256, 0, stream>>>(x, Wbf, pc, w_att, b_att, scores);
    softmax_kernel<<<BB, 256, 0, stream>>>(scores, weights);
    wsum_kernel<<<dim3(DD / 256, SS / 256, BB), 256, 0, stream>>>(x, weights, partial);
    reduce_kernel<<<(BB * DD) / 256, 256, 0, stream>>>(partial, out);
}
